// StrutureAwareHCN_64364379897915
// MI455X (gfx1250) — compile-verified, matrix operand order checked by reading, not run on hardware
//
#include <hip/hip_runtime.h>
#include <hip/hip_bf16.h>
#include <math.h>

typedef __attribute__((ext_vector_type(16))) __bf16 v16bf;
typedef __attribute__((ext_vector_type(8)))  float  v8f;

#define B_   8
#define S_   2500
#define E_   128
#define F_   128
#define L_   8921
#define SP   2528      // padded S (x rows, wf_fs cols, LDS width; 79*32)
#define SSF  2512      // wf_sf rows (157*16)
#define NT_S 157       // s tiles (16 wide)
#define NT_L 558       // l tiles (16 wide)
#define LP   8928      // padded L

// workspace layout (bytes)
#define OFF_XPAD 0
#define SZ_XPAD  (B_*SP*E_*2)
#define OFF_WFSF (OFF_XPAD + SZ_XPAD)
#define SZ_WFSF  (B_*SSF*F_*2)
#define OFF_WFFS (OFF_WFSF + SZ_WFSF)
#define SZ_WFFS  (B_*F_*SP*2)
#define OFF_WT   (OFF_WFFS + SZ_WFFS)
#define SZ_WT    (17*F_*E_*2)
#define OFF_WU   (OFF_WT + SZ_WT)

// Load a 16x32 bf16 fragment in the CDNA5 "A" VGPR pattern from a row-major
// matrix (also used for B by loading B^T). ISA 7.12.2: lanes 0-15 = row,
// lanes 16-31 = same rows, K offset +8 (VGPR0-3) / +8 in upper half (VGPR4-7).
__device__ __forceinline__ v16bf load_frag16(const __bf16* base, int ld) {
  int lane = threadIdx.x & 31;
  int row  = lane & 15;
  int hi   = lane >> 4;
  const __bf16* r = base + (size_t)row * ld;
  union { unsigned int u[8]; v16bf v; } U;
#pragma unroll
  for (int p = 0; p < 8; ++p) {
    int col = (p < 4) ? (2*p + 8*hi) : (2*p + 8 + 8*hi);  // p>=4: 16+2*(p-4)
    U.u[p] = *(const unsigned int*)(r + col);
  }
  return U.v;
}

// -------------------- kernel 1: embedding gather + halo pad --------------------
__global__ void __launch_bounds__(128) embed_kernel(const int* __restrict__ docs,
                                                    const float* __restrict__ embed_w,
                                                    __bf16* __restrict__ x_pad) {
  int r = blockIdx.x % SP;
  int b = blockIdx.x / SP;
  int e = threadIdx.x;
  int s = r - 4;                               // 4-row halo (max pad for K=9)
  float v = 0.f;
  if (s >= 0 && s < S_) {
    int tok = docs[b * S_ + s];
    v = embed_w[(size_t)tok * E_ + e];
  }
  x_pad[((size_t)b * SP + r) * E_ + e] = (__bf16)v;
}

// ---------------- kernel 2: weight repack (conv taps + wu) to bf16 ----------------
__global__ void __launch_bounds__(256) prep_kernel(const float* __restrict__ w3,
                                                   const float* __restrict__ w5,
                                                   const float* __restrict__ w9,
                                                   const float* __restrict__ word_U,
                                                   __bf16* __restrict__ wt,
                                                   __bf16* __restrict__ wu) {
  const int N1 = 17 * F_ * E_;
  const int N2 = LP * F_;
  for (int idx = blockIdx.x * blockDim.x + threadIdx.x; idx < N1 + N2;
       idx += gridDim.x * blockDim.x) {
    if (idx < N1) {
      int tg  = idx / (F_ * E_);
      int rem = idx % (F_ * E_);
      float v;
      if (tg < 3)      v = w3[rem * 3 + tg];
      else if (tg < 8) v = w5[rem * 5 + (tg - 3)];
      else             v = w9[rem * 9 + (tg - 8)];
      wt[idx] = (__bf16)v;
    } else {
      int j = idx - N1;
      int l = j / F_;
      wu[j] = (__bf16)((l < L_) ? word_U[j] : 0.f);
    }
  }
}

// ------------- kernel 3: 3x conv1d (WMMA) + tanh + max, dual-layout out -------------
__global__ void __launch_bounds__(256) conv_kernel(const __bf16* __restrict__ x_pad,
                                                   const __bf16* __restrict__ wt,
                                                   const float* __restrict__ b3,
                                                   const float* __restrict__ b5,
                                                   const float* __restrict__ b9,
                                                   __bf16* __restrict__ wf_sf,
                                                   __bf16* __restrict__ wf_fs) {
  int b    = blockIdx.x / NT_S;
  int st   = blockIdx.x % NT_S;
  int tid  = threadIdx.x;
  int wave = tid >> 5;
  int lane = tid & 31;
  int row  = lane & 15;
  int hi   = lane >> 4;
  int f_base = wave * 16;
  const __bf16* xb = x_pad + (size_t)b * SP * E_;

  float wmax[8];
#pragma unroll
  for (int i = 0; i < 8; ++i) wmax[i] = -1e30f;

  const int    Ks[3]     = {3, 5, 9};
  const int    shifts[3] = {3, 2, 0};          // 4 - K/2 (halo alignment)
  const int    wtoff[3]  = {0, 3, 8};
  const float* biases[3] = {b3, b5, b9};

#pragma unroll
  for (int cv = 0; cv < 3; ++cv) {
    v8f acc;
#pragma unroll
    for (int i = 0; i < 8; ++i) acc[i] = biases[cv][f_base + i + 8 * hi];
    const __bf16* w0 = wt + (size_t)wtoff[cv] * F_ * E_;
    int K = Ks[cv], sh = shifts[cv];
    for (int t = 0; t < K; ++t) {
      const __bf16* brow = xb + (size_t)(st * 16 + t + sh) * E_;
      const __bf16* arow = w0 + (size_t)t * F_ * E_ + (size_t)f_base * E_;
#pragma unroll
      for (int ec = 0; ec < 4; ++ec) {
        v16bf A  = load_frag16(arow + ec * 32, E_);
        v16bf Bv = load_frag16(brow + ec * 32, E_);
        acc = __builtin_amdgcn_wmma_f32_16x16x32_bf16(false, A, false, Bv,
                                                      (short)0, acc, false, false);
      }
    }
#pragma unroll
    for (int i = 0; i < 8; ++i) wmax[i] = fmaxf(wmax[i], tanhf(acc[i]));
  }

  int s = st * 16 + row;
  if (s < S_) {
#pragma unroll
    for (int i = 0; i < 8; ++i) {
      int f = f_base + i + 8 * hi;
      __bf16 v = (__bf16)wmax[i];
      wf_fs[(size_t)b * F_ * SP + (size_t)f * SP + s] = v;
      wf_sf[(size_t)b * SSF * F_ + (size_t)s * F_ + f] = v;
    }
  }
}

// ------ kernel 4: label attention: scores(WMMA)->LDS softmax->context(WMMA)->y ------
__global__ void __launch_bounds__(256) attn_kernel(const __bf16* __restrict__ wf_sf,
                                                   const __bf16* __restrict__ wf_fs,
                                                   const __bf16* __restrict__ wu,
                                                   const float* __restrict__ final_w,
                                                   const float* __restrict__ final_b,
                                                   float* __restrict__ y_out,
                                                   float* __restrict__ ctx_out) {
  int b    = blockIdx.x / NT_L;
  int lt   = blockIdx.x % NT_L;
  int tid  = threadIdx.x;
  int wave = tid >> 5;
  int lane = tid & 31;
  int row  = lane & 15;
  int hi   = lane >> 4;

  extern __shared__ char smem[];
  float*  sc    = (float*)smem;                                   // 16 x SP f32
  __bf16* al    = (__bf16*)(smem + 16 * SP * sizeof(float));      // 16 x SP bf16
  float*  ypart = (float*)(smem + 16 * SP * sizeof(float) + 16 * SP * sizeof(__bf16));

  // init score padding columns [SP-16, SP) and y partials
  { int r = tid >> 4, c = SP - 16 + (tid & 15); sc[r * SP + c] = -1e30f; }
  if (tid < 16) ypart[tid] = 0.f;

  // preload wu A-fragments for this label tile (K = F = 128 -> 4 chunks)
  const __bf16* wuB = wu + (size_t)lt * 16 * F_;
  v16bf wuf[4];
#pragma unroll
  for (int fc = 0; fc < 4; ++fc) wuf[fc] = load_frag16(wuB + fc * 32, F_);

  // ---- phase 1: scores[l, s] into LDS, s-tiles striped across 8 waves ----
  const __bf16* sfB = wf_sf + (size_t)b * SSF * F_;
  for (int st = wave; st < NT_S; st += 8) {
    v8f acc;
#pragma unroll
    for (int i = 0; i < 8; ++i) acc[i] = 0.f;
#pragma unroll
    for (int fc = 0; fc < 4; ++fc) {
      v16bf Bv = load_frag16(sfB + (size_t)(st * 16) * F_ + fc * 32, F_);
      acc = __builtin_amdgcn_wmma_f32_16x16x32_bf16(false, wuf[fc], false, Bv,
                                                    (short)0, acc, false, false);
    }
    int s = st * 16 + row;
#pragma unroll
    for (int i = 0; i < 8; ++i) {
      int r = i + 8 * hi;
      sc[r * SP + s] = (s < S_) ? acc[i] : -1e30f;
    }
  }
  __syncthreads();

  // ---- phase 2: row softmax, normalized alpha stored as bf16 ----
  for (int r = wave; r < 16; r += 8) {
    float m = -1e30f;
    for (int c = lane; c < SP; c += 32) m = fmaxf(m, sc[r * SP + c]);
#pragma unroll
    for (int off = 16; off > 0; off >>= 1) m = fmaxf(m, __shfl_xor(m, off, 32));
    float ssum = 0.f;
    for (int c = lane; c < SP; c += 32) {
      float e = __expf(sc[r * SP + c] - m);
      sc[r * SP + c] = e;
      ssum += e;
    }
#pragma unroll
    for (int off = 16; off > 0; off >>= 1) ssum += __shfl_xor(ssum, off, 32);
    float inv = 1.f / ssum;
    for (int c = lane; c < SP; c += 32) al[r * SP + c] = (__bf16)(sc[r * SP + c] * inv);
  }
  __syncthreads();

  // ---- phase 3: context[l, f] = alpha @ wf^T ; each wave owns one 16-f tile ----
  int f_base = wave * 16;
  const __bf16* fsB = wf_fs + (size_t)b * F_ * SP + (size_t)f_base * SP;
  v8f acc;
#pragma unroll
  for (int i = 0; i < 8; ++i) acc[i] = 0.f;
  for (int kc = 0; kc < SP / 32; ++kc) {       // 79 chunks over s
    v16bf A  = load_frag16(al + kc * 32, SP);
    v16bf Bv = load_frag16(fsB + kc * 32, SP);
    acc = __builtin_amdgcn_wmma_f32_16x16x32_bf16(false, A, false, Bv,
                                                  (short)0, acc, false, false);
  }

  int f = f_base + row;
#pragma unroll
  for (int i = 0; i < 8; ++i) {
    int ll = i + 8 * hi;
    int l  = lt * 16 + ll;
    if (l < L_) {
      float v = acc[i];
      ctx_out[((size_t)b * L_ + l) * F_ + f] = v;
      atomicAdd(&ypart[ll], v * final_w[(size_t)l * F_ + f]);
    }
  }
  __syncthreads();
  if (tid < 16) {
    int l = lt * 16 + tid;
    if (l < L_) y_out[(size_t)b * L_ + l] = ypart[tid] + final_b[l];
  }
}

extern "C" void kernel_launch(void* const* d_in, const int* in_sizes, int n_in,
                              void* d_out, int out_size, void* d_ws, size_t ws_size,
                              hipStream_t stream) {
  const int*   docs    = (const int*)  d_in[0];
  const float* embed_w = (const float*)d_in[4];
  const float* word_U  = (const float*)d_in[5];
  const float* final_w = (const float*)d_in[6];
  const float* final_b = (const float*)d_in[7];
  const float* w3      = (const float*)d_in[8];
  const float* b3      = (const float*)d_in[9];
  const float* w5      = (const float*)d_in[10];
  const float* b5      = (const float*)d_in[11];
  const float* w9      = (const float*)d_in[12];
  const float* b9      = (const float*)d_in[13];

  char* ws = (char*)d_ws;
  __bf16* x_pad = (__bf16*)(ws + OFF_XPAD);
  __bf16* wf_sf = (__bf16*)(ws + OFF_WFSF);
  __bf16* wf_fs = (__bf16*)(ws + OFF_WFFS);
  __bf16* wt    = (__bf16*)(ws + OFF_WT);
  __bf16* wu    = (__bf16*)(ws + OFF_WU);

  float* y   = (float*)d_out;
  float* ctx = y + (size_t)B_ * L_;

  hipLaunchKernelGGL(embed_kernel, dim3(B_ * SP), dim3(E_), 0, stream,
                     docs, embed_w, x_pad);
  hipLaunchKernelGGL(prep_kernel, dim3(2048), dim3(256), 0, stream,
                     w3, w5, w9, word_U, wt, wu);
  hipLaunchKernelGGL(conv_kernel, dim3(B_ * NT_S), dim3(256), 0, stream,
                     x_pad, wt, b3, b5, b9, wf_sf, wf_fs);

  size_t lds = 16 * SP * sizeof(float) + 16 * SP * sizeof(__bf16) + 16 * sizeof(float);
  hipFuncSetAttribute((const void*)attn_kernel,
                      hipFuncAttributeMaxDynamicSharedMemorySize, (int)lds);
  hipLaunchKernelGGL(attn_kernel, dim3(B_ * NT_L), dim3(256), lds, stream,
                     wf_sf, wf_fs, wu, final_w, final_b, y, ctx);
}
